// HNNPotential_90683939488427
// MI455X (gfx1250) — matrix operand order repaired and checked by reading, not compiled, not run on hardware
//
#include <hip/hip_runtime.h>
#include <hip/hip_bf16.h>

typedef __attribute__((ext_vector_type(16))) _Float16 v16h;
typedef __attribute__((ext_vector_type(8)))  float    v8f;

#define NPTS   1024
#define HDIM   64
#define NPAIRS 523776      // 1024*1023/2
#define NTILES 32736       // NPAIRS / 16 (exact)
#define NWGX   256
#define NB     4
#define EPS2   1.0e-4f     // EPS*EPS, EPS=0.01

// silu(x) = x * sigmoid(x) = x * rcp(1 + e^-x)
// v_mul (log2e) + v_exp + v_add + v_rcp + v_mul : 5 VALU ops, no IEEE div
// sequence (v_div_scale/v_div_fmas/v_div_fixup).  rcp is ~1 ulp, far below
// the f16 precision of the WMMA inputs.  Saturates correctly: x<<0 -> e^-x
// = inf -> rcp = 0 -> silu = 0 ; x>>0 -> rcp(1) = 1 -> silu = x.
__device__ __forceinline__ float silu_f(float x) {
  return x * __builtin_amdgcn_rcpf(1.0f + __expf(-x));
}

__global__ __launch_bounds__(256) void hnn_main_kernel(
    const float* __restrict__ pos, const float* __restrict__ W1,
    const float* __restrict__ b1,  const float* __restrict__ W2,
    const float* __restrict__ b2,  const float* __restrict__ W3,
    float* __restrict__ ws) {
  __shared__ float    s_pos[NPTS * 3];     // 12 KB: this batch's positions
  __shared__ _Float16 s_w2[HDIM * HDIM];   // 8 KB:  W2 in f16
  __shared__ float    s_red[256];

  const int tid = threadIdx.x;
  const int b   = blockIdx.y;

  // Stage pos[b] and f16(W2) into LDS cooperatively.
  for (int idx = tid; idx < NPTS * 3; idx += 256)
    s_pos[idx] = pos[b * NPTS * 3 + idx];
  for (int idx = tid; idx < HDIM * HDIM; idx += 256)
    s_w2[idx] = (_Float16)W2[idx];
  __syncthreads();

  const int lane = tid & 31;
  const int wave = tid >> 5;
  const int col  = lane & 15;          // N index within 16-wide tile / M row
  const int hiH  = (lane & 16) ? 1 : 0;

  // Per-lane W1/b1 in A-fragment element order.
  // A f16 16x32 layout: element j of fragment <-> k = kbase + j + (j&8) + 8*hiH
  float w1v[32], b1v[32];
#pragma unroll
  for (int f = 0; f < 2; ++f) {
#pragma unroll
    for (int jj = 0; jj < 16; ++jj) {
      const int k = f * 32 + jj + (jj & 8) + hiH * 8;
      w1v[f * 16 + jj] = W1[k];
      b1v[f * 16 + jj] = b1[k];
    }
  }
  // Per-lane epilogue constants: n = 16*t + col
  float b2v[4], w3v[4];
#pragma unroll
  for (int t = 0; t < 4; ++t) {
    b2v[t] = b2[t * 16 + col];
    w3v[t] = W3[t * 16 + col];
  }
  // B fragments (persistent in VGPRs): B f16 32x16 layout:
  // element j <-> k = 32*h + j + (j&8) + 8*hiH ; column n = 16*t + col
  v16h bf[4][2];
#pragma unroll
  for (int t = 0; t < 4; ++t) {
#pragma unroll
    for (int h = 0; h < 2; ++h) {
#pragma unroll
      for (int jj = 0; jj < 16; ++jj) {
        bf[t][h][jj] = s_w2[(h * 32 + jj + (jj & 8) + hiH * 8) * HDIM + t * 16 + col];
      }
    }
  }

  float acc = 0.0f;
  const int gw     = blockIdx.x * 8 + wave;  // global wave id within batch
  const int stride = NWGX * 8;

  for (int tile = gw; tile < NTILES; tile += stride) {
    const int p = tile * 16 + col;           // linear pair index (i<j, triu k=1)
    // Closed-form decode of (i,j): S(i) = i*(2047-i)/2, then integer fixup.
    float rr = sqrtf(fmaxf(2047.0f * 2047.0f - 8.0f * (float)p, 0.0f));
    int i = (int)floorf((2047.0f - rr) * 0.5f);
    i = i < 0 ? 0 : (i > 1022 ? 1022 : i);
    while (i > 0 && (i * (2047 - i)) / 2 > p) --i;
    while (((i + 1) * (2046 - i)) / 2 <= p) ++i;
    const int j = p - (i * (2047 - i)) / 2 + i + 1;

    const float dx = s_pos[i * 3 + 0] - s_pos[j * 3 + 0];
    const float dy = s_pos[i * 3 + 1] - s_pos[j * 3 + 1];
    const float dz = s_pos[i * 3 + 2] - s_pos[j * 3 + 2];
    const float d  = sqrtf(fmaf(dx, dx, fmaf(dy, dy, fmaf(dz, dz, EPS2))));

    // Layer 1 directly into A fragments (each lane computes exactly the h1
    // elements its WMMA A-fragment slots require).
    v16h a0, a1;
#pragma unroll
    for (int jj = 0; jj < 16; ++jj) {
      a0[jj] = (_Float16)silu_f(fmaf(d, w1v[jj],      b1v[jj]));
      a1[jj] = (_Float16)silu_f(fmaf(d, w1v[16 + jj], b1v[16 + jj]));
    }

    // Layer 2 GEMM: 4 N-tiles x 2 K-steps of v_wmma_f32_16x16x32_f16,
    // fused with layer-3 dot (summed over all rows -> per-lane scalar acc).
#pragma unroll
    for (int t = 0; t < 4; ++t) {
      v8f c = {0.f, 0.f, 0.f, 0.f, 0.f, 0.f, 0.f, 0.f};
      c = __builtin_amdgcn_wmma_f32_16x16x32_f16(false, a0, false, bf[t][0],
                                                 (short)0, c, false, false);
      c = __builtin_amdgcn_wmma_f32_16x16x32_f16(false, a1, false, bf[t][1],
                                                 (short)0, c, false, false);
#pragma unroll
      for (int r = 0; r < 8; ++r)
        acc = fmaf(silu_f(c[r] + b2v[t]), w3v[t], acc);
    }
  }

  // Deterministic block reduction -> one partial per (batch, block).
  s_red[tid] = acc;
  __syncthreads();
  for (int off = 128; off > 0; off >>= 1) {
    if (tid < off) s_red[tid] += s_red[tid + off];
    __syncthreads();
  }
  if (tid == 0) ws[b * NWGX + blockIdx.x] = s_red[0];
}

__global__ void hnn_final_kernel(const float* __restrict__ ws,
                                 const float* __restrict__ b3,
                                 float* __restrict__ out) {
  const int b = threadIdx.x;
  if (b < NB) {
    float s = 0.0f;
    for (int k = 0; k < NWGX; ++k) s += ws[b * NWGX + k];  // fixed order
    out[b] = (s + (float)NPAIRS * b3[0]) * (1.0f / (float)NPTS);
  }
}

extern "C" void kernel_launch(void* const* d_in, const int* in_sizes, int n_in,
                              void* d_out, int out_size, void* d_ws, size_t ws_size,
                              hipStream_t stream) {
  const float* pos = (const float*)d_in[0];
  const float* W1  = (const float*)d_in[1];
  const float* b1  = (const float*)d_in[2];
  const float* W2  = (const float*)d_in[3];
  const float* b2  = (const float*)d_in[4];
  const float* W3  = (const float*)d_in[5];
  const float* b3  = (const float*)d_in[6];
  float* out = (float*)d_out;
  float* ws  = (float*)d_ws;   // needs NB*NWGX*4 = 4 KB

  hipLaunchKernelGGL(hnn_main_kernel, dim3(NWGX, NB), dim3(256), 0, stream,
                     pos, W1, b1, W2, b2, W3, ws);
  hipLaunchKernelGGL(hnn_final_kernel, dim3(1), dim3(32), 0, stream,
                     ws, b3, out);
}
